// RNN_39788577030725
// MI455X (gfx1250) — compile-verified, matrix operand order checked by reading, not run on hardware
//
#include <hip/hip_runtime.h>
#include <math.h>

#define HID   512
#define TSEQ  512
#define VOCAB 25
#define NKB   16      // K blocks of 32 over K=512
#define LSTR  520     // padded LDS row stride (ushorts) to dodge bank conflicts
#define WCKB  8       // recurrent-weight K-blocks cached in LDS (8/16 = 256KB)
#define WCUSH (32 * WCKB * 512)          // cached weight ushorts (131072 = 256KB)

typedef __attribute__((ext_vector_type(16))) __bf16 v16bf;
typedef __attribute__((ext_vector_type(8)))  float  v8f;

struct B32 { unsigned int d[8]; };   // 32 bytes: one bf16 WMMA fragment per lane

__device__ __forceinline__ v8f vzero8() {
    v8f z;
    #pragma unroll
    for (int i = 0; i < 8; i++) z[i] = 0.0f;
    return z;
}

__device__ __forceinline__ unsigned short f2bf(float f) {
    unsigned int u = __builtin_bit_cast(unsigned int, f);
    u += 0x7FFFu + ((u >> 16) & 1u);           // round-to-nearest-even
    return (unsigned short)(u >> 16);
}

// K offset inside a 32-wide K block for dword j (0..7) and lane group g (0/1),
// per CDNA5 16-bit A-matrix layout (B mirrored with N across lanes).
__device__ __forceinline__ int kmap(int j, int g) {
    return ((j < 4) ? 0 : 16) + g * 8 + (j & 3) * 2;
}

__device__ __forceinline__ v8f wmma_bf16(v16bf a, v16bf b, v8f c) {
    return __builtin_amdgcn_wmma_f32_16x16x32_bf16(false, a, false, b,
                                                   (short)0, c, false, false);
}

// Load one packed B fragment (32B contiguous per lane). Works for global or
// LDS-derived pointers; address space is inferred after inlining.
__device__ __forceinline__ v16bf load_bfrag(const unsigned short* p, int lane) {
    B32 v = *(const B32*)(p + lane * 16);
    return __builtin_bit_cast(v16bf, v);
}

// Load an A fragment (16x32 bf16) from an LDS row-major [16][stride] buffer.
__device__ __forceinline__ v16bf load_afrag_lds(const unsigned short* sh, int stride,
                                                int kb, int lane) {
    int m = lane & 15, g = lane >> 4;
    const unsigned short* row = sh + m * stride + kb * 32;
    B32 v;
    #pragma unroll
    for (int j = 0; j < 8; j++)
        v.d[j] = *(const unsigned int*)(row + kmap(j, g));
    return __builtin_bit_cast(v16bf, v);
}

// Load an A fragment from a global row-major bf16 matrix (row stride in ushorts).
__device__ __forceinline__ v16bf load_afrag_g(const unsigned short* __restrict__ base,
                                              int rowstride, int kb, int lane) {
    int m = lane & 15, g = lane >> 4;
    const unsigned short* row = base + (size_t)m * rowstride + kb * 32;
    B32 v;
    #pragma unroll
    for (int j = 0; j < 8; j++)
        v.d[j] = *(const unsigned int*)(row + kmap(j, g));
    return __builtin_bit_cast(v16bf, v);
}

// ---------------------------------------------------------------------------
// Pack W (realN x 512 f32, row-major) into WMMA B fragments, bf16.
// dst fragment (nt,kb): 512 ushorts; lane-major, 16 values/lane.
// B[k][n] = W[n][k]; rows n >= realN are zero-padded.
// ---------------------------------------------------------------------------
__global__ void pack_b(const float* __restrict__ W, unsigned short* __restrict__ dst,
                       int ntiles, int realN) {
    int idx = blockIdx.x * blockDim.x + threadIdx.x;   // one dword (2 bf16) per thread
    if (idx >= (ntiles << 12)) return;
    int j    = idx & 7;
    int lane = (idx >> 3) & 31;
    int kb   = (idx >> 8) & 15;
    int nt   = idx >> 12;
    int g    = lane >> 4;
    int n    = nt * 16 + (lane & 15);
    int k0   = kb * 32 + kmap(j, g);
    unsigned int lo = 0, hi = 0;
    if (n < realN) {
        lo = f2bf(W[(size_t)n * HID + k0]);
        hi = f2bf(W[(size_t)n * HID + k0 + 1]);
    }
    unsigned int* d32 = (unsigned int*)dst;
    d32[((((nt * NKB + kb) << 9) + (lane << 4)) >> 1) + j] = lo | (hi << 16);
}

// ---------------------------------------------------------------------------
// Recurrent scan. One block = 16 batch rows; 8 waves x 4 N-tiles = N=512.
// h double-buffered in LDS (bf16); K-blocks 0..WCKB-1 of the recurrent weights
// are held resident in LDS (256KB) to halve per-step L2 traffic.
// mode 0: xt = embed(W_ih0, tokens)+biases, writes full h sequence.
// mode 1: xt = xw1[b][t][n], writes only last h.
// ---------------------------------------------------------------------------
__global__ __launch_bounds__(256) void rnn_scan(
    const unsigned short* __restrict__ Wpk,   // packed recurrent weights (32 ntiles)
    const float* __restrict__ bias0,          // b_ih (mode0) / b_hh (mode1)
    const float* __restrict__ bias1,          // b_hh (mode0) / nullptr
    const int*   __restrict__ tokens,         // mode0 only: (B, T) int32
    const float* __restrict__ Wemb,           // mode0 only: (HID, VOCAB) f32
    const float* __restrict__ xin,            // mode1 only: (B, T, HID) f32
    unsigned short* __restrict__ hseq,        // mode0 only: (B, T, HID) bf16
    unsigned short* __restrict__ hlast,       // mode1 only: (B, HID) bf16
    int mode)
{
    // 256KB weight cache + 2 x (16 x LSTR) h buffers = ~295KB of the 320KB WGP LDS
    __shared__ __align__(16) unsigned short smem[WCUSH + 2 * 16 * LSTR];
    unsigned short* hb = smem + WCUSH;

    const int tid  = threadIdx.x;
    const int lane = tid & 31;
    const int w    = tid >> 5;
    const int bg   = blockIdx.x;              // batch group (16 rows)
    const int g    = lane >> 4;
    const int mcol = lane & 15;

    // Fill LDS weight cache: fragments (nt, kb<WCKB), layout (nt*WCKB+kb)<<9.
    {
        const uint4* src = (const uint4*)Wpk;
        uint4*       dst = (uint4*)smem;
        for (int i = tid; i < (WCUSH / 8); i += 256) {
            int frag = i >> 6, within = i & 63;
            int nt = frag >> 3, kb = frag & 7;          // frag = nt*WCKB + kb
            dst[i] = src[((nt * NKB + kb) << 6) + within];
        }
    }
    for (int i = tid; i < 16 * LSTR; i += 256) hb[i] = 0;   // h(t=-1) = 0
    __syncthreads();

    float bias[4];
    int   ncol[4];
    #pragma unroll
    for (int q = 0; q < 4; q++) {
        int n   = (w * 4 + q) * 16 + mcol;
        ncol[q] = n;
        float b = bias0[n];
        if (bias1) b += bias1[n];
        bias[q] = b;
    }

    for (int t = 0; t < TSEQ; t++) {
        const int cur = t & 1, nxt = cur ^ 1;
        unsigned short* hcur = hb + cur * 16 * LSTR;
        unsigned short* hnxt = hb + nxt * 16 * LSTR;

        // Issue the per-step additive input loads first so they overlap WMMAs.
        float xt[4][8];
        if (mode == 0) {
            int tk[8];
            #pragma unroll
            for (int r = 0; r < 8; r++)
                tk[r] = tokens[(size_t)(bg * 16 + g * 8 + r) * TSEQ + t];
            #pragma unroll
            for (int q = 0; q < 4; q++)
                #pragma unroll
                for (int r = 0; r < 8; r++)
                    xt[q][r] = Wemb[(size_t)ncol[q] * VOCAB + tk[r]];
        } else {
            #pragma unroll
            for (int q = 0; q < 4; q++)
                #pragma unroll
                for (int r = 0; r < 8; r++) {
                    size_t row = (size_t)(bg * 16 + g * 8 + r) * TSEQ + t;
                    xt[q][r] = xin[(row << 9) + ncol[q]];
                }
        }

        v8f acc[4];
        #pragma unroll
        for (int q = 0; q < 4; q++) acc[q] = vzero8();

        // K-blocks whose weights stream from L2 (issue global loads early).
        for (int kb = WCKB; kb < NKB; kb++) {
            v16bf a = load_afrag_lds(hcur, LSTR, kb, lane);
            #pragma unroll
            for (int q = 0; q < 4; q++) {
                v16bf b = load_bfrag(Wpk + (((w * 4 + q) * NKB + kb) << 9), lane);
                acc[q] = wmma_bf16(a, b, acc[q]);
            }
        }
        // K-blocks whose weights are LDS-resident.
        for (int kb = 0; kb < WCKB; kb++) {
            v16bf a = load_afrag_lds(hcur, LSTR, kb, lane);
            #pragma unroll
            for (int q = 0; q < 4; q++) {
                v16bf b = load_bfrag(smem + (((w * 4 + q) * WCKB + kb) << 9), lane);
                acc[q] = wmma_bf16(a, b, acc[q]);
            }
        }

        #pragma unroll
        for (int q = 0; q < 4; q++) {
            #pragma unroll
            for (int r = 0; r < 8; r++) {
                float v = tanhf(acc[q][r] + bias[q] + xt[q][r]);
                unsigned short hv = f2bf(v);
                int mm = g * 8 + r;
                hnxt[mm * LSTR + ncol[q]] = hv;
                if (mode == 0) {
                    size_t row = (size_t)(bg * 16 + mm) * TSEQ + t;
                    hseq[(row << 9) + ncol[q]] = hv;
                } else if (t == TSEQ - 1) {
                    hlast[(size_t)(bg * 16 + mm) * HID + ncol[q]] = hv;
                }
            }
        }
        __syncthreads();
    }
}

// ---------------------------------------------------------------------------
// xw1 = h1 @ W_ih1^T + b_ih1 :  (32768 x 512) = (32768 x 512)bf16 x packed B
// One block (4 waves) per M-tile; wave covers 8 N-tiles.
// ---------------------------------------------------------------------------
__global__ __launch_bounds__(128) void gemm_xw1(
    const unsigned short* __restrict__ A,    // (B*T, HID) bf16
    const unsigned short* __restrict__ Wpk,  // packed W_ih1
    const float* __restrict__ bias,
    float* __restrict__ out)                 // (B*T, HID) f32
{
    const int lane = threadIdx.x & 31;
    const int w    = threadIdx.x >> 5;
    const int mt   = blockIdx.x;
    const int g    = lane >> 4;
    const int mcol = lane & 15;

    v8f acc[8];
    #pragma unroll
    for (int q = 0; q < 8; q++) acc[q] = vzero8();

    const unsigned short* arow = A + (size_t)mt * 16 * HID;
    for (int kb = 0; kb < NKB; kb++) {
        v16bf a = load_afrag_g(arow, HID, kb, lane);
        #pragma unroll
        for (int q = 0; q < 8; q++) {
            v16bf b = load_bfrag(Wpk + (((w * 8 + q) * NKB + kb) << 9), lane);
            acc[q] = wmma_bf16(a, b, acc[q]);
        }
    }
    #pragma unroll
    for (int q = 0; q < 8; q++) {
        int n = (w * 8 + q) * 16 + mcol;
        float bn = bias[n];
        #pragma unroll
        for (int r = 0; r < 8; r++) {
            size_t mm = (size_t)mt * 16 + g * 8 + r;
            out[mm * HID + n] = acc[q][r] + bn;
        }
    }
}

// ---------------------------------------------------------------------------
// out = h2_last @ W_out^T + b_out : (64 x 1000). One wave per (ntile, mtile).
// ---------------------------------------------------------------------------
__global__ __launch_bounds__(32) void gemm_out(
    const unsigned short* __restrict__ A,    // (64, HID) bf16
    const unsigned short* __restrict__ Wpk,  // packed W_out (63 ntiles, zero-padded)
    const float* __restrict__ bias,
    float* __restrict__ out)                 // (64, 1000) f32
{
    const int lane = threadIdx.x & 31;
    const int nt   = blockIdx.x;             // 0..62
    const int mt   = blockIdx.y;             // 0..3
    const int g    = lane >> 4;

    v8f acc = vzero8();
    const unsigned short* arow = A + (size_t)mt * 16 * HID;
    for (int kb = 0; kb < NKB; kb++) {
        v16bf a = load_afrag_g(arow, HID, kb, lane);
        v16bf b = load_bfrag(Wpk + ((nt * NKB + kb) << 9), lane);
        acc = wmma_bf16(a, b, acc);
    }
    int n = nt * 16 + (lane & 15);
    if (n < 1000) {
        float bn = bias[n];
        #pragma unroll
        for (int r = 0; r < 8; r++) {
            int mm = mt * 16 + g * 8 + r;
            out[(size_t)mm * 1000 + n] = acc[r] + bn;
        }
    }
}

extern "C" void kernel_launch(void* const* d_in, const int* in_sizes, int n_in,
                              void* d_out, int out_size, void* d_ws, size_t ws_size,
                              hipStream_t stream) {
    const int*   x     = (const int*)d_in[0];
    const float* W_ih0 = (const float*)d_in[1];
    const float* W_hh0 = (const float*)d_in[2];
    const float* b_ih0 = (const float*)d_in[3];
    const float* b_hh0 = (const float*)d_in[4];
    const float* W_ih1 = (const float*)d_in[5];
    const float* W_hh1 = (const float*)d_in[6];
    const float* b_ih1 = (const float*)d_in[7];
    const float* b_hh1 = (const float*)d_in[8];
    const float* W_out = (const float*)d_in[9];
    const float* b_out = (const float*)d_in[10];
    (void)in_sizes; (void)n_in; (void)out_size; (void)ws_size;

    char* ws = (char*)d_ws;
    size_t off = 0;
    auto take = [&](size_t bytes) -> char* {
        char* p = ws + off;
        off = (off + bytes + 255) & ~(size_t)255;
        return p;
    };
    unsigned short* pkWhh0 = (unsigned short*)take((size_t)32 * NKB * 512 * 2); // 512KB
    unsigned short* pkWih1 = (unsigned short*)take((size_t)32 * NKB * 512 * 2);
    unsigned short* pkWhh1 = (unsigned short*)take((size_t)32 * NKB * 512 * 2);
    unsigned short* pkWout = (unsigned short*)take((size_t)63 * NKB * 512 * 2);
    unsigned short* h1seq  = (unsigned short*)take((size_t)64 * TSEQ * HID * 2); // 32MB
    float*          xw1    = (float*)take((size_t)64 * TSEQ * HID * 4);          // 64MB
    unsigned short* h2last = (unsigned short*)take((size_t)64 * HID * 2);

    // 1) Pack weights into WMMA-fragment order (bf16).
    pack_b<<<512, 256, 0, stream>>>(W_hh0, pkWhh0, 32, 512);
    pack_b<<<512, 256, 0, stream>>>(W_ih1, pkWih1, 32, 512);
    pack_b<<<512, 256, 0, stream>>>(W_hh1, pkWhh1, 32, 512);
    pack_b<<<1008, 256, 0, stream>>>(W_out, pkWout, 63, 1000);

    // 2) Layer-0 scan: xt = embed + b_ih0 + b_hh0, h in LDS, write h1 sequence.
    rnn_scan<<<4, 256, 0, stream>>>(pkWhh0, b_ih0, b_hh0, x, W_ih0,
                                    nullptr, h1seq, nullptr, 0);

    // 3) xw1 = h1 @ W_ih1^T + b_ih1 (parallel GEMM, M = 32768).
    gemm_xw1<<<2048, 128, 0, stream>>>(h1seq, pkWih1, b_ih1, xw1);

    // 4) Layer-1 scan: xt = xw1 (+ b_hh1), keep only final state.
    rnn_scan<<<4, 256, 0, stream>>>(pkWhh1, b_hh1, nullptr, nullptr, nullptr,
                                    xw1, nullptr, h2last, 1);

    // 5) out = h2_last @ W_out^T + b_out.
    gemm_out<<<dim3(63, 4), 32, 0, stream>>>(h2last, pkWout, b_out, (float*)d_out);
}
